// HyperGraphAttentionLayerSparse_15229954031624
// MI455X (gfx1250) — matrix operand order repaired and checked by reading, not compiled, not run on hardware
//
#include <hip/hip_runtime.h>
#include <hip/hip_bf16.h>

// ---------------------------------------------------------------------------
// HyperGraph attention layer, CDNA5 (gfx1250) WMMA implementation, round 4.
// - b128-contiguous fragment loads/stores everywhere
// - per-output GEMM orientation (compute D^T when dest is row-major)
// - dual-tile / dual-chain processing so independent WMMAs hide load latency
//   (breaks the compiler's single-register-set load->wait->wmma serialization)
// - async global->LDS staging (ASYNCcnt) for scores1 and adj tiles
// ---------------------------------------------------------------------------

#define BB   128
#define NN   512            // N1 == N2 == 512
#define DD   128
#define NEGV (-1e9f)
#define INV_TEMP 0.08838834764831845f   // 1/sqrt(128)

typedef __bf16 bf16_t;
typedef __attribute__((ext_vector_type(16))) __bf16 v16bf;
typedef __attribute__((ext_vector_type(8)))  __bf16 v8bf;
typedef __attribute__((ext_vector_type(4)))  __bf16 v4bf;
typedef __attribute__((ext_vector_type(8)))  float  v8f;
typedef __attribute__((ext_vector_type(4)))  float  v4f;

__device__ inline v16bf frag_cat(v8bf lo, v8bf hi) {
    return __builtin_shufflevector(lo, hi, 0,1,2,3,4,5,6,7,8,9,10,11,12,13,14,15);
}

// ---------------- WMMA fragment helpers (ISA 7.12.2 layouts) ----------------
__device__ inline v16bf load_a_frag(const bf16_t* A, int lda, int row0, int k0, int lane) {
    int row = row0 + (lane & 15);
    int kh  = (lane >> 4) * 8;
    const bf16_t* p = A + (size_t)row * lda + k0 + kh;
    v8bf lo = *(const v8bf*)(p);
    v8bf hi = *(const v8bf*)(p + 16);
    return frag_cat(lo, hi);
}

__device__ inline v16bf load_a_frag_f32(const float* A, int lda, int row0, int k0, int lane) {
    int row = row0 + (lane & 15);
    int kh  = (lane >> 4) * 8;
    const float* p = A + (size_t)row * lda + k0 + kh;
    v16bf a;
#pragma unroll
    for (int i = 0; i < 8; ++i) a[i]     = (bf16_t)p[i];
#pragma unroll
    for (int i = 0; i < 8; ++i) a[8 + i] = (bf16_t)p[16 + i];
    return a;
}

__device__ inline v16bf load_bt_frag(const bf16_t* M, int ldm, int k0, int col0, int lane) {
    int col = col0 + (lane & 15);
    int kh  = (lane >> 4) * 8;
    const bf16_t* p = M + (size_t)col * ldm + k0 + kh;
    v8bf lo = *(const v8bf*)(p);
    v8bf hi = *(const v8bf*)(p + 16);
    return frag_cat(lo, hi);
}

__device__ inline v16bf load_bt_frag_f32(const float* M, int ldm, int k0, int col0, int lane) {
    int col = col0 + (lane & 15);
    int kh  = (lane >> 4) * 8;
    const float* p = M + (size_t)col * ldm + k0 + kh;
    v16bf b;
#pragma unroll
    for (int j = 0; j < 16; ++j) b[j] = (bf16_t)p[j < 8 ? j : j + 8];
    return b;
}

__device__ inline void store_d_f32(float* C, int ldc, int row0, int col0, int lane, v8f d) {
    int col = col0 + (lane & 15);
    int rb  = row0 + (lane >> 4) * 8;
#pragma unroll
    for (int i = 0; i < 8; ++i) C[(size_t)(rb + i) * ldc + col] = d[i];
}

__device__ inline void store_d_bf16(bf16_t* C, int ldc, int row0, int col0, int lane, v8f d) {
    int col = col0 + (lane & 15);
    int rb  = row0 + (lane >> 4) * 8;
#pragma unroll
    for (int i = 0; i < 8; ++i) C[(size_t)(rb + i) * ldc + col] = (bf16_t)d[i];
}

__device__ inline void store_dT_bf16(bf16_t* CT, int ldct, int row0, int col0, int lane, v8f d) {
    int col = col0 + (lane & 15);
    int rb  = row0 + (lane >> 4) * 8;
    v8bf o;
#pragma unroll
    for (int i = 0; i < 8; ++i) o[i] = (bf16_t)d[i];
    *(v8bf*)(CT + (size_t)col * ldct + rb) = o;
}

__device__ inline void store_dT_relu_f32(float* CT, int ldct, int row0, int col0, int lane, v8f d) {
    int col = col0 + (lane & 15);
    int rb  = row0 + (lane >> 4) * 8;
    float* p = CT + (size_t)col * ldct + rb;
    v4f lo = { fmaxf(d[0],0.f), fmaxf(d[1],0.f), fmaxf(d[2],0.f), fmaxf(d[3],0.f) };
    v4f hi = { fmaxf(d[4],0.f), fmaxf(d[5],0.f), fmaxf(d[6],0.f), fmaxf(d[7],0.f) };
    *(v4f*)(p)     = lo;
    *(v4f*)(p + 4) = hi;
}

__device__ inline v8f wmma_bf16(v16bf a, v16bf b, v8f c) {
    return __builtin_amdgcn_wmma_f32_16x16x32_bf16(false, a, false, b, (short)0, c, false, false);
}

__device__ inline float wave_max32(float v) {
#pragma unroll
    for (int off = 16; off > 0; off >>= 1) v = fmaxf(v, __shfl_xor(v, off, 32));
    return v;
}
__device__ inline float wave_sum32(float v) {
#pragma unroll
    for (int off = 16; off > 0; off >>= 1) v += __shfl_xor(v, off, 32);
    return v;
}

// ------------- async global -> LDS staging (ISA 15.18.3, ASYNCcnt) ----------
__device__ inline void async_copy_b128(void* lds_ptr, const void* gaddr) {
    unsigned loff = (unsigned)(size_t)lds_ptr;   // low 32 bits = LDS offset
    asm volatile("global_load_async_to_lds_b128 %0, %1, off"
                 :: "v"(loff), "v"(gaddr) : "memory");
}
__device__ inline void wait_async0() {
    asm volatile("s_wait_asynccnt 0" ::: "memory");
}

// ---------------------------------------------------------------------------
// weights -> bf16 transposed (W^T[f][d] = W[d][f])
// ---------------------------------------------------------------------------
__global__ void k_cvt_wT(const float* __restrict__ src, bf16_t* __restrict__ dstT) {
    int i = blockIdx.x * blockDim.x + threadIdx.x;   // i = d*DD + f
    if (i >= DD * DD) return;
    int d = i >> 7, f = i & 127;
    dstT[(size_t)f * DD + d] = (bf16_t)src[i];
}

// ---------------------------------------------------------------------------
// Kernel 1: per (b, 16-node tile), both projections FUSED in one tile loop:
//   x4att[n][f] = (W2^T . x^T)^T   swapped  (chain-pipelined w2T A-frags)
//   xtT[f][n]   = (x . W)^T        normal
// Two independent accumulate chains per iteration hide operand-load latency.
// grid = B*(NN/16), block = 32
// ---------------------------------------------------------------------------
__global__ void k_xproj(const float* __restrict__ x,
                        const bf16_t* __restrict__ w2T,
                        const bf16_t* __restrict__ wT,
                        bf16_t* __restrict__ x4att,   // [B][N][D] row-major
                        bf16_t* __restrict__ xtT) {   // [B][D][N]
    int b  = blockIdx.x >> 5;
    int nt = blockIdx.x & 31;
    int lane = threadIdx.x;
    const float* X = x + (size_t)b * NN * DD;
    bf16_t* o4 = x4att + (size_t)b * NN * DD;
    bf16_t* oT = xtT   + (size_t)b * DD * NN;

    v16bf bx[4];   // B' = x rows (f32->bf16) for the swapped GEMM
    v16bf ax[4];   // A  = x rows for the normal GEMM
#pragma unroll
    for (int ks = 0; ks < 4; ++ks) {
        bx[ks] = load_bt_frag_f32(X, DD, ks * 32, nt * 16, lane);
        ax[ks] = load_a_frag_f32 (X, DD, nt * 16, ks * 32, lane);
    }

    v16bf aw = load_a_frag(w2T, DD, 0, 0, lane);   // chained w2T fragments
#pragma unroll
    for (int t = 0; t < 8; ++t) {
        v8f acc4 = {}, acct = {};
#pragma unroll
        for (int ks = 0; ks < 4; ++ks) {
            int tn = t * 4 + ks + 1;               // next (ft,ks) in sequence
            v16bf awn = (tn < 32)
                ? load_a_frag(w2T, DD, (tn >> 2) * 16, (tn & 3) * 32, lane) : aw;
            v16bf bw = load_bt_frag(wT, DD, ks * 32, t * 16, lane);
            acc4 = wmma_bf16(aw, bx[ks], acc4);    // chain 1 (swapped)
            acct = wmma_bf16(ax[ks], bw, acct);    // chain 2 (normal)
            aw = awn;
        }
        store_dT_bf16(o4, DD, t * 16, nt * 16, lane, acc4);   // x4att[n][f], b128
        store_dT_bf16(oT, NN, nt * 16, t * 16, lane, acct);   // xtT[f][n],   b128
    }
}

// ---------------------------------------------------------------------------
// Kernel 2: scores1[b,n] = (wc . x_4att[b,n]) / sqrt(D)
// ---------------------------------------------------------------------------
__global__ void k_scores1(const bf16_t* __restrict__ x4att,
                          const float* __restrict__ wc,
                          float* __restrict__ s1) {
    int idx = blockIdx.x * blockDim.x + threadIdx.x;   // b*NN + n
    if (idx >= BB * NN) return;
    const v8bf* p = (const v8bf*)(x4att + (size_t)idx * DD);
    float s = 0.f;
#pragma unroll
    for (int c = 0; c < DD / 8; ++c) {
        v8bf v = p[c];
#pragma unroll
        for (int j = 0; j < 8; ++j) s += wc[c * 8 + j] * (float)v[j];
    }
    s1[idx] = s * INV_TEMP;
}

// ---------------------------------------------------------------------------
// Kernel 3 (edge): per (b, 16-edge tile): masked softmax of scores1 against
// adj rows, then edge computed SWAPPED: D[f][e] = sum_n xtT[f][n] * P[e][n].
// Final GEMM runs TWO f-tiles per pass: each LDS B'-fragment feeds 2 WMMAs,
// two chained A-fragment streams keep loads in flight.
// grid = B*(NN/16), block = 32
// ---------------------------------------------------------------------------
__global__ void k_edge(const float* __restrict__ s1g,
                       const int*   __restrict__ adj,
                       const bf16_t* __restrict__ xtT,
                       float* __restrict__ edge_out,   // relu(edge) [B,N1,D]
                       bf16_t* __restrict__ edge_bf,   // [B][N1][D] row-major
                       bf16_t* __restrict__ edgeT) {   // [B][D][N1]
    __shared__ __align__(16) float  s1[NN];
    __shared__ __align__(16) bf16_t pbuf[16 * NN];     // P[e][n]

    int b  = blockIdx.x >> 5;
    int et = blockIdx.x & 31;
    int lane = threadIdx.x;

    // async-stage scores1 row (2KB) into LDS
    {
        const float* gs = s1g + (size_t)b * NN;
        for (int t = lane; t < NN / 4; t += 32)
            async_copy_b128(&s1[t * 4], gs + t * 4);
        wait_async0();
    }
    __syncthreads();

    for (int e = 0; e < 16; ++e) {
        const int* arow = adj + ((size_t)b * NN + et * 16 + e) * NN;
        float m = -3.4e38f;
        for (int i0 = lane * 4; i0 < NN; i0 += 128) {
            int4 a4 = *(const int4*)(arow + i0);
            float v0 = a4.x ? s1[i0 + 0] : NEGV;
            float v1 = a4.y ? s1[i0 + 1] : NEGV;
            float v2 = a4.z ? s1[i0 + 2] : NEGV;
            float v3 = a4.w ? s1[i0 + 3] : NEGV;
            m = fmaxf(m, fmaxf(fmaxf(v0, v1), fmaxf(v2, v3)));
        }
        m = wave_max32(m);
        float sum = 0.f;
        for (int i0 = lane * 4; i0 < NN; i0 += 128) {
            int4 a4 = *(const int4*)(arow + i0);
            sum += __expf((a4.x ? s1[i0 + 0] : NEGV) - m);
            sum += __expf((a4.y ? s1[i0 + 1] : NEGV) - m);
            sum += __expf((a4.z ? s1[i0 + 2] : NEGV) - m);
            sum += __expf((a4.w ? s1[i0 + 3] : NEGV) - m);
        }
        sum = wave_sum32(sum);
        float inv = 1.0f / sum;
        for (int i0 = lane * 4; i0 < NN; i0 += 128) {
            int4 a4 = *(const int4*)(arow + i0);
            v4bf o;
            o[0] = (bf16_t)(__expf((a4.x ? s1[i0 + 0] : NEGV) - m) * inv);
            o[1] = (bf16_t)(__expf((a4.y ? s1[i0 + 1] : NEGV) - m) * inv);
            o[2] = (bf16_t)(__expf((a4.z ? s1[i0 + 2] : NEGV) - m) * inv);
            o[3] = (bf16_t)(__expf((a4.w ? s1[i0 + 3] : NEGV) - m) * inv);
            *(v4bf*)(&pbuf[e * NN + i0]) = o;
        }
    }
    __syncthreads();

    // swapped GEMM, two f-tiles per pass
    const bf16_t* AT = xtT + (size_t)b * DD * NN;
    float*  oR = edge_out + (size_t)b * NN * DD;
    bf16_t* oB = edge_bf  + (size_t)b * NN * DD;
    bf16_t* oT = edgeT    + (size_t)b * DD * NN;
#pragma unroll
    for (int fp = 0; fp < 4; ++fp) {
        int f0 = fp * 32;
        v8f acc0 = {}, acc1 = {};
        v16bf a0 = load_a_frag(AT, NN, f0,      0, lane);
        v16bf a1 = load_a_frag(AT, NN, f0 + 16, 0, lane);
        for (int ks = 0; ks < 16; ++ks) {
            v16bf a0n = (ks < 15) ? load_a_frag(AT, NN, f0,      (ks + 1) * 32, lane) : a0;
            v16bf a1n = (ks < 15) ? load_a_frag(AT, NN, f0 + 16, (ks + 1) * 32, lane) : a1;
            v16bf bb  = load_bt_frag(pbuf, NN, ks * 32, 0, lane);
            acc0 = wmma_bf16(a0, bb, acc0);
            acc1 = wmma_bf16(a1, bb, acc1);
            a0 = a0n; a1 = a1n;
        }
        store_dT_relu_f32(oR, DD, f0,      et * 16, lane, acc0);
        store_dT_relu_f32(oR, DD, f0 + 16, et * 16, lane, acc1);
        store_dT_bf16   (oB, DD, f0,      et * 16, lane, acc0);
        store_dT_bf16   (oB, DD, f0 + 16, et * 16, lane, acc1);
        store_d_bf16    (oT, NN, f0,      et * 16, lane, acc0);   // edgeT[f][e], strided
        store_d_bf16    (oT, NN, f0 + 16, et * 16, lane, acc1);
    }
}

// ---------------------------------------------------------------------------
// Kernel 4: e4att[e][f] via SWAPPED GEMM, TWO e-tiles per wave:
//   D[f][e] = sum_d W3T[f][d] * edge[e][d]; each W3T A-frag feeds 2 WMMAs.
// grid = B*(NN/32), block = 32
// ---------------------------------------------------------------------------
__global__ void k_e4att(const bf16_t* __restrict__ edge_bf,
                        const bf16_t* __restrict__ w3T,
                        bf16_t* __restrict__ e4att) {   // [B][N1][D] row-major
    int b  = blockIdx.x >> 4;
    int ep = blockIdx.x & 15;          // pair of e-tiles
    int e0 = ep * 32;
    int lane = threadIdx.x;
    const bf16_t* E = edge_bf + (size_t)b * NN * DD;
    bf16_t* o = e4att + (size_t)b * NN * DD;

    v16bf be0[4], be1[4];              // B' = edge rows for the two e-tiles
#pragma unroll
    for (int ks = 0; ks < 4; ++ks) {
        be0[ks] = load_bt_frag(E, DD, ks * 32, e0,      lane);
        be1[ks] = load_bt_frag(E, DD, ks * 32, e0 + 16, lane);
    }

    v16bf a = load_a_frag(w3T, DD, 0, 0, lane);   // chained weight fragments
#pragma unroll
    for (int ft = 0; ft < 8; ++ft) {
        v8f acc0 = {}, acc1 = {};
#pragma unroll
        for (int ks = 0; ks < 4; ++ks) {
            int tn = ft * 4 + ks + 1;
            v16bf an = (tn < 32)
                ? load_a_frag(w3T, DD, (tn >> 2) * 16, (tn & 3) * 32, lane) : a;
            acc0 = wmma_bf16(a, be0[ks], acc0);
            acc1 = wmma_bf16(a, be1[ks], acc1);
            a = an;
        }
        store_dT_bf16(o, DD, ft * 16, e0,      lane, acc0);   // e4att[e][f], b128
        store_dT_bf16(o, DD, ft * 16, e0 + 16, lane, acc1);
    }
}

// ---------------------------------------------------------------------------
// Kernel 5 (node): per (b, 16-node tile):
//   S[n][e] = x4att @ e4att^T (normal, B'-frags double-buffered) -> LDS
//   adj^T tile async-staged; masked softmax over edges -> P2[n][e]
//   node SWAPPED, two f-tiles per pass: D[f][n] = sum_e edgeT[f][e]*P2[n][e]
// grid = B*(NN/16), block = 32
// ---------------------------------------------------------------------------
__global__ void k_node(const bf16_t* __restrict__ x4att,
                       const bf16_t* __restrict__ e4att,
                       const bf16_t* __restrict__ edgeT,
                       const int*   __restrict__ adj,
                       float* __restrict__ node_out) {
    __shared__ __align__(16) float  sbuf[16 * NN];   // S[n-local][e]
    __shared__ __align__(16) bf16_t pbuf[16 * NN];   // P2[n-local][e]
    __shared__ __align__(16) int    aint[NN * 16];   // adj[b][e][node tile]

    int b  = blockIdx.x >> 5;
    int nt = blockIdx.x & 31;
    int lane = threadIdx.x;

    // async-stage adj^T tile (512 x 16 ints = 32KB)
    {
        const int* ab = adj + (size_t)b * NN * NN + nt * 16;
        for (int t = lane; t < NN * 4; t += 32) {
            int e = t >> 2, c = t & 3;
            async_copy_b128(&aint[e * 16 + c * 4], ab + (size_t)e * NN + c * 4);
        }
    }

    const bf16_t* A  = x4att + (size_t)b * NN * DD;
    const bf16_t* E4 = e4att + (size_t)b * NN * DD;
    v16bf afr[4];
#pragma unroll
    for (int ks = 0; ks < 4; ++ks) afr[ks] = load_a_frag(A, DD, nt * 16, ks * 32, lane);

    // S = A @ e4att^T, B'-fragments double-buffered across e-tiles
    {
        v16bf bfr[4];
#pragma unroll
        for (int ks = 0; ks < 4; ++ks) bfr[ks] = load_bt_frag(E4, DD, ks * 32, 0, lane);
        for (int et = 0; et < 32; ++et) {
            v16bf bnx[4];
            if (et < 31) {
#pragma unroll
                for (int ks = 0; ks < 4; ++ks)
                    bnx[ks] = load_bt_frag(E4, DD, ks * 32, (et + 1) * 16, lane);
            }
            v8f acc = {};
#pragma unroll
            for (int ks = 0; ks < 4; ++ks) acc = wmma_bf16(afr[ks], bfr[ks], acc);
            store_d_f32(sbuf, NN, 0, et * 16, lane, acc);
            if (et < 31) {
#pragma unroll
                for (int ks = 0; ks < 4; ++ks) bfr[ks] = bnx[ks];
            }
        }
    }
    wait_async0();
    __syncthreads();

    // masked softmax over edges, scale 1/sqrt(D)
    for (int r = 0; r < 16; ++r) {
        float m = -3.4e38f;
        for (int e = lane; e < NN; e += 32) {
            float v = aint[e * 16 + r] ? sbuf[r * NN + e] * INV_TEMP : NEGV;
            m = fmaxf(m, v);
        }
        m = wave_max32(m);
        float sum = 0.f;
        for (int e = lane; e < NN; e += 32) {
            float v = aint[e * 16 + r] ? sbuf[r * NN + e] * INV_TEMP : NEGV;
            sum += __expf(v - m);
        }
        sum = wave_sum32(sum);
        float inv = 1.0f / sum;
        for (int e = lane; e < NN; e += 32) {
            float v = aint[e * 16 + r] ? sbuf[r * NN + e] * INV_TEMP : NEGV;
            pbuf[r * NN + e] = (bf16_t)(__expf(v - m) * inv);
        }
    }
    __syncthreads();

    // node (swapped), two f-tiles per pass
    const bf16_t* AT = edgeT + (size_t)b * DD * NN;
    float* o = node_out + (size_t)b * NN * DD;
#pragma unroll
    for (int fp = 0; fp < 4; ++fp) {
        int f0 = fp * 32;
        v8f acc0 = {}, acc1 = {};
        v16bf a0 = load_a_frag(AT, NN, f0,      0, lane);
        v16bf a1 = load_a_frag(AT, NN, f0 + 16, 0, lane);
        for (int ks = 0; ks < 16; ++ks) {
            v16bf a0n = (ks < 15) ? load_a_frag(AT, NN, f0,      (ks + 1) * 32, lane) : a0;
            v16bf a1n = (ks < 15) ? load_a_frag(AT, NN, f0 + 16, (ks + 1) * 32, lane) : a1;
            v16bf bb  = load_bt_frag(pbuf, NN, ks * 32, 0, lane);
            acc0 = wmma_bf16(a0, bb, acc0);
            acc1 = wmma_bf16(a1, bb, acc1);
            a0 = a0n; a1 = a1n;
        }
        store_dT_relu_f32(o, DD, f0,      nt * 16, lane, acc0);   // node[n][f], 2xb128
        store_dT_relu_f32(o, DD, f0 + 16, nt * 16, lane, acc1);
    }
}

// ---------------------------------------------------------------------------
// Host-side launch
// ---------------------------------------------------------------------------
extern "C" void kernel_launch(void* const* d_in, const int* in_sizes, int n_in,
                              void* d_out, int out_size, void* d_ws, size_t ws_size,
                              hipStream_t stream) {
    const float* x   = (const float*)d_in[0];   // [B,N2,D]
    const int*   adj = (const int*)  d_in[1];   // [B,N1,N2]
    const float* w   = (const float*)d_in[2];   // [D,D]
    const float* w2  = (const float*)d_in[3];   // [D,D]
    const float* w3  = (const float*)d_in[4];   // [D,D]
    const float* wc  = (const float*)d_in[5];   // [D]

    float* node_out = (float*)d_out;                            // [B,N2,D]
    float* edge_out = (float*)d_out + (size_t)BB * NN * DD;     // [B,N1,D]

    char* ws = (char*)d_ws;
    const size_t BND2 = (size_t)BB * NN * DD * sizeof(bf16_t);  // 16 MiB each
    bf16_t* x4att  = (bf16_t*)ws;  ws += BND2;   // [B][N][D]
    bf16_t* xtT    = (bf16_t*)ws;  ws += BND2;   // [B][D][N]
    bf16_t* edgeb  = (bf16_t*)ws;  ws += BND2;   // [B][N][D]
    bf16_t* edgeT  = (bf16_t*)ws;  ws += BND2;   // [B][D][N]
    bf16_t* e4att  = (bf16_t*)ws;  ws += BND2;   // [B][N][D]
    bf16_t* wT     = (bf16_t*)ws;  ws += (size_t)DD * DD * sizeof(bf16_t);
    bf16_t* w2T    = (bf16_t*)ws;  ws += (size_t)DD * DD * sizeof(bf16_t);
    bf16_t* w3T    = (bf16_t*)ws;  ws += (size_t)DD * DD * sizeof(bf16_t);
    float*  s1     = (float*)ws;   ws += (size_t)BB * NN * sizeof(float);

    k_cvt_wT<<<(DD * DD + 255) / 256, 256, 0, stream>>>(w,  wT);
    k_cvt_wT<<<(DD * DD + 255) / 256, 256, 0, stream>>>(w2, w2T);
    k_cvt_wT<<<(DD * DD + 255) / 256, 256, 0, stream>>>(w3, w3T);

    const int tiles = BB * (NN / 16);   // 4096 wave-tiles

    k_xproj<<<tiles, 32, 0, stream>>>(x, w2T, wT, x4att, xtT);
    k_scores1<<<(BB * NN + 255) / 256, 256, 0, stream>>>(x4att, wc, s1);
    k_edge<<<tiles, 32, 0, stream>>>(s1, adj, xtT, edge_out, edgeb, edgeT);
    k_e4att<<<BB * (NN / 32), 32, 0, stream>>>(edgeb, w3T, e4att);
    k_node<<<tiles, 32, 0, stream>>>(x4att, e4att, edgeT, adj, node_out);
}
